// AttentionSuper_2525440770362
// MI455X (gfx1250) — matrix operand order repaired and checked by reading, not compiled, not run on hardware
//
#include <hip/hip_runtime.h>

#define BB 64
#define NN 197
#define HH 12
#define NPJ 224   // j padded to multiple of 32 for the PV K-dim

typedef _Float16 f16;
typedef __attribute__((ext_vector_type(4)))  _Float16 v4h;
typedef __attribute__((ext_vector_type(8)))  _Float16 v8h;
typedef __attribute__((ext_vector_type(16))) _Float16 v16h;
typedef __attribute__((ext_vector_type(8)))  float    v8f;

// ---- dynamic LDS layout (bytes), all 16B aligned, total 298496 < 320KB ----
#define OFF_SW    0u          // per-wave S/P: 8 * 16*224 * 4   = 114688
#define OFF_WB    114688u     // per-wave bins f32: 8 * 16*64*4 = 32768
#define OFF_PVH   147456u     // pv||ph f16: 208 * 64 * 2       = 26624
#define OFF_QS    174080u     // Q f16: 208 * 72 * 2            = 29952
#define OFF_KS    204032u     // K f16: 208 * 72 * 2            = 29952
#define OFF_VT    233984u     // V^T f16: 64 * 232 * 2          = 29696
#define OFF_TK    263680u     // [tk_v;tk_h] f16: 64 * 72 * 2   = 9216
#define OFF_TV    272896u     // [tv_v;tv_h]^T f16: 64 * 72 * 2 = 9216
#define OFF_BH    282112u     // per-wave bins f16: 8*16*64*2   = 16384
#define SMEM_BYTES 298496u

// A fragment (16x32 f16): lane r=lane&15 holds row r; halves = K[hs*8..+8) and K[16+hs*8..+8)
__device__ __forceinline__ v16h ld_a(const f16* p, int rs, int row, int k0, int hs) {
  const f16* q = p + row * rs + k0 + hs * 8;
  union { v16h v; v8h h[2]; } u;
  u.h[0] = *(const v8h*)(q);
  u.h[1] = *(const v8h*)(q + 16);
  return u.v;
}
// B fragment (32x16 f16): lane c=lane&15 holds col c; halves = K[hs*8..+8) and K[16+hs*8..+8)
// (source stored so that the K dimension is contiguous per column)
__device__ __forceinline__ v16h ld_b(const f16* p, int rs, int col0, int c, int k0, int hs) {
  const f16* q = p + (col0 + c) * rs + k0 + hs * 8;
  union { v16h v; v8h h[2]; } u;
  u.h[0] = *(const v8h*)(q);
  u.h[1] = *(const v8h*)(q + 16);
  return u.v;
}
#define AMDWMMA(a, b, acc) \
  __builtin_amdgcn_wmma_f32_16x16x32_f16(false, (a), false, (b), (short)0, (acc), false, false)

// relative-position table indices (cls token at i==0 / j==0 -> padded index 0)
__device__ __forceinline__ void relidx(int i, int j, int& iv, int& ih) {
  if (i == 0 || j == 0) { iv = 0; ih = 0; return; }
  int ri = i - 1, rj = j - 1;
  int dv = rj / 14 - ri / 14;
  int dh = rj % 14 - ri % 14;
  dv = dv < -14 ? -14 : (dv > 14 ? 14 : dv);
  dh = dh < -14 ? -14 : (dh > 14 ? 14 : dh);
  iv = dv + 15; ih = dh + 15;
}

__device__ __forceinline__ float wred_max(float v) {
#pragma unroll
  for (int o = 16; o > 0; o >>= 1) v = fmaxf(v, __shfl_xor(v, o, 32));
  return v;
}
__device__ __forceinline__ float wred_sum(float v) {
#pragma unroll
  for (int o = 16; o > 0; o >>= 1) v += __shfl_xor(v, o, 32);
  return v;
}

__global__ __launch_bounds__(256, 1)
void attn_rel_kernel(const float* __restrict__ x,
                     const float* __restrict__ tkv, const float* __restrict__ tkh,
                     const float* __restrict__ tvv, const float* __restrict__ tvh,
                     float* __restrict__ out) {
  extern __shared__ char smem[];
  float* Sw  = (float*)(smem + OFF_SW);
  float* Wb  = (float*)(smem + OFF_WB);
  f16*   PVH = (f16*)(smem + OFF_PVH);
  f16*   Qs  = (f16*)(smem + OFF_QS);
  f16*   Ks  = (f16*)(smem + OFF_KS);
  f16*   VT  = (f16*)(smem + OFF_VT);
  f16*   Tk  = (f16*)(smem + OFF_TK);
  f16*   Tv  = (f16*)(smem + OFF_TV);
  f16*   Bh  = (f16*)(smem + OFF_BH);

  const int tid = threadIdx.x;
  const int b = blockIdx.x / HH, h = blockIdx.x % HH;
  const float* xb = x + (size_t)b * NN * 3072 + h * 64;

  // ---------- Phase A: stage Q,K (row-major f16), V^T, tables ----------
  for (int q = tid; q < NN * 16; q += 256) {
    int n = q >> 4, c4 = q & 15;
    const float* row = xb + (size_t)n * 3072 + c4 * 4;
    float4 fq = *(const float4*)(row);
    float4 fk = *(const float4*)(row + 768);
    float4 fv = *(const float4*)(row + 1536);
    v4h hq = { (f16)fq.x, (f16)fq.y, (f16)fq.z, (f16)fq.w };
    v4h hk = { (f16)fk.x, (f16)fk.y, (f16)fk.z, (f16)fk.w };
    *(v4h*)(Qs + n * 72 + c4 * 4) = hq;
    *(v4h*)(Ks + n * 72 + c4 * 4) = hk;
    int d0 = c4 * 4;
    VT[(d0 + 0) * 232 + n] = (f16)fv.x;
    VT[(d0 + 1) * 232 + n] = (f16)fv.y;
    VT[(d0 + 2) * 232 + n] = (f16)fv.z;
    VT[(d0 + 3) * 232 + n] = (f16)fv.w;
  }
  // zero-pad Q/K rows 197..207, V^T columns 197..231
  for (int q = tid; q < 11 * 72; q += 256) { Qs[NN * 72 + q] = (f16)0.f; Ks[NN * 72 + q] = (f16)0.f; }
  for (int q = tid; q < 64 * 35; q += 256) { int d = q / 35, cc = NN + q % 35; VT[d * 232 + cc] = (f16)0.f; }
  // Tk rows 0..29 = tk_v, rows 32..61 = tk_h (K-dim = table row, contiguous per t)
  // Tv transposed: Tv[d][t] so B-fragment K chunks are contiguous
  for (int q = tid; q < 30 * 64; q += 256) {
    int t = q >> 6, d = q & 63;
    Tk[t * 72 + d]        = (f16)tkv[q];
    Tk[(32 + t) * 72 + d] = (f16)tkh[q];
    Tv[d * 72 + t]        = (f16)tvv[t * 64 + d];
    Tv[d * 72 + 32 + t]   = (f16)tvh[t * 64 + d];
  }
  for (int q = tid; q < 4 * 72; q += 256) {       // zero Tk rows 30,31,62,63
    int rsel = q / 72, cc = q % 72; int rr = (rsel < 2) ? (30 + rsel) : (60 + rsel);
    Tk[rr * 72 + cc] = (f16)0.f;
  }
  for (int q = tid; q < 64 * 4; q += 256) {       // zero Tv cols 30,31,62,63
    int d = q >> 2, ssel = q & 3; int cc = (ssel < 2) ? (30 + ssel) : (60 + ssel);
    Tv[d * 72 + cc] = (f16)0.f;
  }
  __syncthreads();  // only barrier in the kernel; waves are independent below

  const int wave = tid >> 5, lane = tid & 31;
  const int c = lane & 15, hs = lane >> 4;
  float* SwW = Sw + wave * 16 * 224;     // per-wave 16x224 f32 scores
  f16*   PwW = (f16*)SwW;                // reused in-place as 16x224 f16 probs
  float* WbW = Wb + wave * 16 * 64;      // per-wave bins (wv || wh) f32
  f16*   BhW = Bh + wave * 16 * 64;      // bins f16 for WMMA

  for (int it = wave; it < 13; it += 8) {
    const int i0 = it * 16;

    // (1) pv||ph = Q_tile @ [tk_v;tk_h]^T   (16x64 result into PVH)
#pragma unroll
    for (int ct = 0; ct < 4; ct++) {
      v8f acc = {0.f, 0.f, 0.f, 0.f, 0.f, 0.f, 0.f, 0.f};
#pragma unroll
      for (int k0 = 0; k0 < 64; k0 += 32)
        acc = AMDWMMA(ld_a(Qs, 72, i0 + c, k0, hs), ld_b(Tk, 72, ct * 16, c, k0, hs), acc);
#pragma unroll
      for (int g = 0; g < 8; g++)
        PVH[(i0 + g + 8 * hs) * 64 + ct * 16 + c] = (f16)acc[g];
    }

    // (2) S = Q_tile @ K^T
    for (int jt = 0; jt < 13; jt++) {
      v8f acc = {0.f, 0.f, 0.f, 0.f, 0.f, 0.f, 0.f, 0.f};
#pragma unroll
      for (int k0 = 0; k0 < 64; k0 += 32)
        acc = AMDWMMA(ld_a(Qs, 72, i0 + c, k0, hs), ld_b(Ks, 72, jt * 16, c, k0, hs), acc);
#pragma unroll
      for (int g = 0; g < 8; g++)
        SwW[(g + 8 * hs) * 224 + jt * 16 + c] = acc[g];
    }

    for (int q = lane; q < 16 * 64; q += 32) WbW[q] = 0.f;
    asm volatile("" ::: "memory");

    // (3) rel-bias add + softmax + probability histogram, one row per pass
    for (int m = 0; m < 16; m++) {
      const int i = i0 + m;
      float* Srow = SwW + m * 224;
      float mx = -3.0e38f;
      for (int j = lane; j < NN; j += 32) {
        int iv, ih; relidx(i, j, iv, ih);
        float s = (Srow[j] + (float)PVH[i * 64 + iv] + (float)PVH[i * 64 + 32 + ih]) * 0.125f;
        Srow[j] = s;
        mx = fmaxf(mx, s);
      }
      mx = wred_max(mx);
      float sm = 0.f;
      for (int j = lane; j < NN; j += 32) {
        float p = __expf(Srow[j] - mx);
        sm += p; Srow[j] = p;
      }
      sm = wred_sum(sm);
      const float inv = 1.f / sm;
      f16* Prow = (f16*)Srow;
      float pr[7]; int ivr[7], ihr[7];
      int cnt = 0;
      for (int j = lane; j < NPJ; j += 32) {         // stage reads first (P aliases S)
        float p = 0.f; int iv = 0, ih = 0;
        if (j < NN) { p = Srow[j] * inv; relidx(i, j, iv, ih); }
        pr[cnt] = p; ivr[cnt] = iv; ihr[cnt] = ih; cnt++;
      }
      asm volatile("" ::: "memory");
      cnt = 0;
      for (int j = lane; j < NPJ; j += 32) {         // then write f16 P + bins
        Prow[j] = (f16)pr[cnt];
        if (j < NN) {
          atomicAdd(&WbW[(m << 6) + ivr[cnt]], pr[cnt]);
          atomicAdd(&WbW[(m << 6) + 32 + ihr[cnt]], pr[cnt]);
        }
        cnt++;
      }
    }
    asm volatile("" ::: "memory");
    for (int q = lane; q < 16 * 64; q += 32) BhW[q] = (f16)WbW[q];
    asm volatile("" ::: "memory");

    // (4) O = P @ V  +  bins @ [tv_v;tv_h]
#pragma unroll
    for (int dt = 0; dt < 4; dt++) {
      v8f acc = {0.f, 0.f, 0.f, 0.f, 0.f, 0.f, 0.f, 0.f};
      for (int jt = 0; jt < 7; jt++)
        acc = AMDWMMA(ld_a(PwW, 448, c, jt * 32, hs), ld_b(VT, 232, dt * 16, c, jt * 32, hs), acc);
#pragma unroll
      for (int k0 = 0; k0 < 64; k0 += 32)
        acc = AMDWMMA(ld_a(BhW, 64, c, k0, hs), ld_b(Tv, 72, dt * 16, c, k0, hs), acc);
#pragma unroll
      for (int g = 0; g < 8; g++) {
        int i = i0 + g + 8 * hs;
        if (i < NN)
          out[((size_t)(b * NN + i)) * 1024 + h * 64 + dt * 16 + c] = acc[g];
      }
    }
    asm volatile("" ::: "memory");
  }
}

__global__ void zero_pad_kernel(float* __restrict__ out) {
  int idx = blockIdx.x * 256 + threadIdx.x;
  if (idx < BB * NN * 256) {
    int bn = idx >> 8, cc = idx & 255;
    out[(size_t)bn * 1024 + 768 + cc] = 0.f;
  }
}

extern "C" void kernel_launch(void* const* d_in, const int* in_sizes, int n_in,
                              void* d_out, int out_size, void* d_ws, size_t ws_size,
                              hipStream_t stream) {
  (void)in_sizes; (void)n_in; (void)d_ws; (void)ws_size; (void)out_size;
  const float* x   = (const float*)d_in[0];
  const float* tkv = (const float*)d_in[1];
  const float* tkh = (const float*)d_in[2];
  const float* tvv = (const float*)d_in[3];
  const float* tvh = (const float*)d_in[4];
  float* out = (float*)d_out;

  hipFuncSetAttribute(reinterpret_cast<const void*>(attn_rel_kernel),
                      hipFuncAttributeMaxDynamicSharedMemorySize, (int)SMEM_BYTES);
  attn_rel_kernel<<<dim3(BB * HH), dim3(256), SMEM_BYTES, stream>>>(x, tkv, tkh, tvv, tvh, out);
  zero_pad_kernel<<<dim3((BB * NN * 256 + 255) / 256), dim3(256), 0, stream>>>(out);
}
